// QModel_24996709663445
// MI455X (gfx1250) — compile-verified, matrix operand order checked by reading, not run on hardware
//
#include <hip/hip_runtime.h>
#include <hip/hip_bf16.h>
#include <cstddef>

// ---------------------------------------------------------------------------
// Quantized LeNet forward for MI455X (gfx1250, wave32, WMMA).
//   conv1+pool fused VALU kernel (K=25, not worth WMMA; avoids 94MB buffer)
//   conv2: explicit LDS im2col + WMMA GEMM + fused requant+pool+flatten
//          (one 512-thread block per image; A-tile 64x512 f16 in LDS)
//   fc1  : WMMA GEMM  M=4096, K=800, N=512(500)
//   fc2+log_softmax: wave-per-row VALU kernel
// All quantized activations are exact small integers -> fp16 is lossless.
// ---------------------------------------------------------------------------

typedef _Float16 h16;
typedef _Float16 v16h __attribute__((ext_vector_type(16)));
typedef _Float16 v8h  __attribute__((ext_vector_type(8)));
typedef float    v8f  __attribute__((ext_vector_type(8)));

#define BATCH 4096

// stats-derived compile-time constants (STATS is fixed in the reference)
#define S_IN_C1    (3.5f/255.0f)     // conv1 input scale
#define INV_S_C1   (255.0f/3.5f)
#define ZP_IN_C1   36.0f             // trunc(0.5*255/3.5)
#define S_IN_C2    (6.0f/255.0f)     // conv2 layer's scale_x
#define S_IN_FC1   (8.0f/255.0f)     // fc1 layer's scale_x
#define S_IN_FC2   (10.0f/255.0f)    // dequant scale before fc2
#define INV_SN_C1  (255.0f/6.0f)     // 1/scale_next, conv1 layer (zp_next=0)
#define INV_SN_C2  (255.0f/8.0f)
#define INV_SN_FC1 (255.0f/10.0f)

__device__ __forceinline__ float clamp255(float x) {
    return fminf(fmaxf(x, 0.0f), 255.0f);
}

// y = mod(round(relu((acc + beff) * inv_scale_next)), 256)
__device__ __forceinline__ float quant_ep(float acc, float beff, float inv_sn) {
    float y = fmaxf((acc + beff) * inv_sn, 0.0f);
    return fmodf(rintf(y), 256.0f);
}

__device__ __forceinline__ void scale_zp(const float* mm, int t, float& s, float& zp) {
    float mn = mm[2 * t], mx = mm[2 * t + 1];
    s = (mx - mn) * (1.0f / 255.0f);
    if (s < 1e-30f) s = 1e-30f;
    zp = truncf(clamp255(-mn / s));
}

// ------------------------- min/max of 6 param tensors ----------------------
__global__ void __launch_bounds__(256)
k_minmax(const float* t0, int n0, const float* t1, int n1,
         const float* t2, int n2, const float* t3, int n3,
         const float* t4, int n4, const float* t5, int n5, float* mm) {
    const float* ptrs[6] = {t0, t1, t2, t3, t4, t5};
    const int ns[6] = {n0, n1, n2, n3, n4, n5};
    int t = blockIdx.x;
    const float* p = ptrs[t];
    int n = ns[t];
    float mn = 3.4e38f, mx = -3.4e38f;
    for (int i = threadIdx.x; i < n; i += blockDim.x) {
        float v = p[i];
        mn = fminf(mn, v);
        mx = fmaxf(mx, v);
    }
    __shared__ float smn[256], smx[256];
    smn[threadIdx.x] = mn; smx[threadIdx.x] = mx;
    __syncthreads();
    for (int s = 128; s > 0; s >>= 1) {
        if ((int)threadIdx.x < s) {
            smn[threadIdx.x] = fminf(smn[threadIdx.x], smn[threadIdx.x + s]);
            smx[threadIdx.x] = fmaxf(smx[threadIdx.x], smx[threadIdx.x + s]);
        }
        __syncthreads();
    }
    if (threadIdx.x == 0) { mm[2 * t] = smn[0]; mm[2 * t + 1] = smx[0]; }
}

// -------------------- effective weights/biases (small) ---------------------
// mm tensor order: 0=conv1_w 1=conv1_b 2=conv2_w 3=conv2_b 4=fc1_w 5=fc1_b
__global__ void __launch_bounds__(256)
k_prep_small(const float* c1w, const float* c1b, const float* c2b, const float* fc1b,
             const float* mm, float* Weff1, float* Beff1, float* Beff2, float* Beff3) {
    int blk = blockIdx.x;
    if (blk == 0) {          // Beff1[20]
        float s, zp; scale_zp(mm, 1, s, zp);
        for (int i = threadIdx.x; i < 20; i += blockDim.x) {
            float bq = rintf(clamp255(zp + c1b[i] / s));
            Beff1[i] = s * (bq + zp);          // '+zp' faithful to source
        }
    } else if (blk == 1) {   // Beff2[64] (pad->0)
        float s, zp; scale_zp(mm, 3, s, zp);
        for (int i = threadIdx.x; i < 64; i += blockDim.x) {
            float v = 0.0f;
            if (i < 50) { float bq = rintf(clamp255(zp + c2b[i] / s)); v = s * (bq + zp); }
            Beff2[i] = v;
        }
    } else if (blk == 2) {   // Beff3[512] (pad->0)
        float s, zp; scale_zp(mm, 5, s, zp);
        for (int i = threadIdx.x; i < 512; i += blockDim.x) {
            float v = 0.0f;
            if (i < 500) { float bq = rintf(clamp255(zp + fc1b[i] / s)); v = s * (bq + zp); }
            Beff3[i] = v;
        }
    } else {                 // Weff1[500] fp32 (conv1 stays VALU)
        float s, zp; scale_zp(mm, 0, s, zp);
        for (int i = threadIdx.x; i < 500; i += blockDim.x) {
            float wq = rintf(clamp255(zp + c1w[i] / s));
            Weff1[i] = S_IN_C1 * s * (wq - zp);
        }
    }
}

// conv2 B-matrix, transposed layout Bt2[n][k], N=64(pad), K=512(pad), fp16
__global__ void __launch_bounds__(256)
k_prep_bt2(const float* c2w, const float* mm, h16* Bt2) {
    int i = blockIdx.x * blockDim.x + threadIdx.x;
    if (i >= 64 * 512) return;
    int n = i >> 9, k = i & 511;
    float v = 0.0f;
    if (n < 50 && k < 500) {
        float s, zp; scale_zp(mm, 2, s, zp);
        float w = c2w[n * 500 + k];            // OIHW flat: k = c*25+kh*5+kw
        float wq = rintf(clamp255(zp + w / s));
        v = S_IN_C2 * s * (wq - zp);
    }
    Bt2[i] = (h16)v;
}

// fc1 B-matrix Bt3[n][k], N=512(pad), K=800, fp16 (fc1_w is already [500][800])
__global__ void __launch_bounds__(256)
k_prep_bt3(const float* fc1w, const float* mm, h16* Bt3) {
    int i = blockIdx.x * blockDim.x + threadIdx.x;
    if (i >= 512 * 800) return;
    int n = i / 800, k = i % 800;
    float v = 0.0f;
    if (n < 500) {
        float s, zp; scale_zp(mm, 4, s, zp);
        float w = fc1w[n * 800 + k];
        float wq = rintf(clamp255(zp + w / s));
        v = S_IN_FC1 * s * (wq - zp);
    }
    Bt3[i] = (h16)v;
}

// ---------------------- input quantization: Xq = q - zp --------------------
__global__ void __launch_bounds__(256)
k_quant_input(const float* x, h16* Xq, int n) {
    int i = blockIdx.x * blockDim.x + threadIdx.x;
    if (i < n) {
        float q = rintf(clamp255(ZP_IN_C1 + x[i] * INV_S_C1));
        Xq[i] = (h16)(q - ZP_IN_C1);
    }
}

// ------------- conv1 (direct, K=25) fused with requant + 2x2 pool ----------
__global__ void __launch_bounds__(256)
k_conv1_pool(const h16* Xq, const float* Weff1, const float* Beff1, h16* P1) {
    __shared__ float sW[500];
    __shared__ float sB[20];
    for (int i = threadIdx.x; i < 500; i += blockDim.x) sW[i] = Weff1[i];
    if (threadIdx.x < 20) sB[threadIdx.x] = Beff1[threadIdx.x];
    __syncthreads();
    int i = blockIdx.x * blockDim.x + threadIdx.x;
    if (i >= BATCH * 144) return;
    int b = i / 144, r = i % 144, py = r / 12, px = r % 12;
    const h16* src = Xq + (size_t)b * 784 + (2 * py) * 28 + 2 * px;
    float patch[36];
#pragma unroll
    for (int dy = 0; dy < 6; dy++)
#pragma unroll
        for (int dx = 0; dx < 6; dx++) patch[dy * 6 + dx] = (float)src[dy * 28 + dx];
    for (int o = 0; o < 20; o++) {
        float a00 = 0.f, a01 = 0.f, a10 = 0.f, a11 = 0.f;
#pragma unroll
        for (int kh = 0; kh < 5; kh++)
#pragma unroll
            for (int kw = 0; kw < 5; kw++) {
                float w = sW[o * 25 + kh * 5 + kw];
                a00 = fmaf(patch[kh * 6 + kw], w, a00);
                a01 = fmaf(patch[kh * 6 + kw + 1], w, a01);
                a10 = fmaf(patch[(kh + 1) * 6 + kw], w, a10);
                a11 = fmaf(patch[(kh + 1) * 6 + kw + 1], w, a11);
            }
        float be = sB[o];
        float q00 = quant_ep(a00, be, INV_SN_C1);
        float q01 = quant_ep(a01, be, INV_SN_C1);
        float q10 = quant_ep(a10, be, INV_SN_C1);
        float q11 = quant_ep(a11, be, INV_SN_C1);
        float m = fmaxf(fmaxf(q00, q01), fmaxf(q10, q11));
        P1[((size_t)b * 20 + o) * 144 + r] = (h16)m;
    }
}

// --------- conv2: LDS im2col + WMMA + fused requant + pool + flatten -------
// One 512-thread block (16 waves, 4x4 wave grid) per image:
//  phase 1: stage 20x12x12 image (5.76KB) + per-k offset table (512 entries,
//           computed once -> no div/mod in any hot loop)
//  phase 2: build explicit im2col tile Atile[64][512] f16 (64KB LDS)
//  phase 3: WMMA GEMM 64x64x512 -- A via ds_load_b128, B via global loads
//  phase 4: requantize into LDS (reuse Atile), 2x2 pool, write P2[b][800]
__global__ void __launch_bounds__(512)
k_conv2_wmma(const h16* P1, const h16* Bt2, const float* Beff2, h16* P2) {
    __shared__ h16 simg[2880];
    __shared__ int kbase[512];
    __shared__ h16 Atile[64 * 512];   // reused as qtile[64][64] in phase 4
    const int b = blockIdx.x;
    const int tid = threadIdx.x;

    // phase 1
    for (int i = tid; i < 2880; i += 512) simg[i] = P1[(size_t)b * 2880 + i];
    {
        int k = tid;                                // blockDim == 512
        int base = -1;
        if (k < 500) {
            int c = k / 25, rem = k % 25;
            base = c * 144 + (rem / 5) * 12 + (rem % 5);
        }
        kbase[k] = base;
    }
    __syncthreads();

    // phase 2: thread owns column k = tid; loops the 64 output pixels
    {
        int base = kbase[tid];
#pragma unroll 8
        for (int r = 0; r < 64; r++) {
            int pix = (r >> 3) * 12 + (r & 7);      // oy*12 + ox
            Atile[r * 512 + tid] = (base >= 0) ? simg[base + pix] : (h16)0;
        }
    }
    __syncthreads();

    // phase 3: GEMM. wave (wm,wn) computes rows wm*16.., cols wn*16..
    const int w = tid >> 5, L = tid & 31;
    const int wm = w & 3, wn = w >> 2;
    const int n0 = wn * 16;
    const int kSel = (L & 16) ? 8 : 0;              // ISA 16-bit frag layout
    const h16* arow = &Atile[(wm * 16 + (L & 15)) * 512];
    const h16* bp = Bt2 + (size_t)(n0 + (L & 15)) * 512;
    v8f acc = {};
    for (int k0 = 0; k0 < 512; k0 += 32) {
        v8h alo = *(const v8h*)(arow + k0 + kSel);
        v8h ahi = *(const v8h*)(arow + k0 + kSel + 16);
        v8h blo = *(const v8h*)(bp + k0 + kSel);
        v8h bhi = *(const v8h*)(bp + k0 + kSel + 16);
        v16h a = __builtin_shufflevector(alo, ahi, 0, 1, 2, 3, 4, 5, 6, 7,
                                         8, 9, 10, 11, 12, 13, 14, 15);
        v16h bb = __builtin_shufflevector(blo, bhi, 0, 1, 2, 3, 4, 5, 6, 7,
                                          8, 9, 10, 11, 12, 13, 14, 15);
        acc = __builtin_amdgcn_wmma_f32_16x16x32_f16(
            false, a, false, bb, (short)0, acc, false, false);
    }
    __syncthreads();   // all Atile reads done before qtile overwrite

    // phase 4a: requantize into qtile[n][r]
    h16* qtile = Atile;
    const int n = n0 + (L & 15);
    if (n < 50) {
        float be = Beff2[n];
#pragma unroll
        for (int v = 0; v < 8; v++) {
            int r = wm * 16 + v + ((L & 16) ? 8 : 0);   // C/D row (7.12.2)
            qtile[n * 64 + r] = (h16)quant_ep(acc[v], be, INV_SN_C2);
        }
    }
    __syncthreads();

    // phase 4b: 2x2 pool + flatten -> P2[b][800], col = c*16 + py*4 + px
    for (int col = tid; col < 800; col += 512) {
        int c = col >> 4, r2 = col & 15, py = r2 >> 2, px = r2 & 3;
        const h16* s = &qtile[c * 64 + (2 * py) * 8 + 2 * px];
        float m = fmaxf(fmaxf((float)s[0], (float)s[1]),
                        fmaxf((float)s[8], (float)s[9]));
        P2[(size_t)b * 800 + col] = (h16)m;
    }
}

// ---------------------- fc1 WMMA GEMM: 4096x800x512 ------------------------
__global__ void __launch_bounds__(256)
k_fc1_wmma(const h16* P2, const h16* Bt3, const float* Beff3, h16* Q3) {
    int w = threadIdx.x >> 5, L = threadIdx.x & 31;
    int wm = w & 3, wn = w >> 2;
    int m0 = blockIdx.x * 64 + wm * 16;
    int n0 = blockIdx.y * 32 + wn * 16;
    int kSel = (L & 16) ? 8 : 0;
    const h16* ap = P2 + (size_t)(m0 + (L & 15)) * 800;
    const h16* bp = Bt3 + (size_t)(n0 + (L & 15)) * 800;
    v8f acc = {};
    for (int k0 = 0; k0 < 800; k0 += 32) {
        v8h alo = *(const v8h*)(ap + k0 + kSel);
        v8h ahi = *(const v8h*)(ap + k0 + kSel + 16);
        v8h blo = *(const v8h*)(bp + k0 + kSel);
        v8h bhi = *(const v8h*)(bp + k0 + kSel + 16);
        v16h a = __builtin_shufflevector(alo, ahi, 0, 1, 2, 3, 4, 5, 6, 7,
                                         8, 9, 10, 11, 12, 13, 14, 15);
        v16h bb = __builtin_shufflevector(blo, bhi, 0, 1, 2, 3, 4, 5, 6, 7,
                                          8, 9, 10, 11, 12, 13, 14, 15);
        acc = __builtin_amdgcn_wmma_f32_16x16x32_f16(
            false, a, false, bb, (short)0, acc, false, false);
    }
    int n = n0 + (L & 15);
    if (n < 500) {
        float be = Beff3[n];
#pragma unroll
        for (int v = 0; v < 8; v++) {
            int row = m0 + v + ((L & 16) ? 8 : 0);
            float q = quant_ep(acc[v], be, INV_SN_FC1);
            Q3[(size_t)row * 512 + n] = (h16)q;
        }
    }
}

// --------------- fc2 (dequant * W^T + b) + log_softmax, wave/row -----------
__global__ void __launch_bounds__(256)
k_fc2_lsm(const h16* Q3, const float* fw, const float* fb, float* out) {
    int L = threadIdx.x & 31, wv = threadIdx.x >> 5;
    int m = blockIdx.x * 8 + wv;
    float acc[10];
#pragma unroll
    for (int j = 0; j < 10; j++) acc[j] = 0.0f;
    for (int k = L; k < 500; k += 32) {
        float xv = S_IN_FC2 * (float)Q3[(size_t)m * 512 + k];
#pragma unroll
        for (int j = 0; j < 10; j++) acc[j] = fmaf(xv, fw[j * 500 + k], acc[j]);
    }
#pragma unroll
    for (int j = 0; j < 10; j++)
        for (int off = 16; off > 0; off >>= 1)
            acc[j] += __shfl_xor(acc[j], off, 32);
    if (L == 0) {
        float l[10], mx = -3.4e38f;
#pragma unroll
        for (int j = 0; j < 10; j++) { l[j] = acc[j] + fb[j]; mx = fmaxf(mx, l[j]); }
        float se = 0.0f;
#pragma unroll
        for (int j = 0; j < 10; j++) se += expf(l[j] - mx);
        float lse = mx + logf(se);
#pragma unroll
        for (int j = 0; j < 10; j++) out[(size_t)m * 10 + j] = l[j] - lse;
    }
}

// ---------------------------------------------------------------------------
extern "C" void kernel_launch(void* const* d_in, const int* in_sizes, int n_in,
                              void* d_out, int out_size, void* d_ws, size_t ws_size,
                              hipStream_t stream) {
    const float* x   = (const float*)d_in[0];
    const float* c1w = (const float*)d_in[1];
    const float* c1b = (const float*)d_in[2];
    const float* c2w = (const float*)d_in[3];
    const float* c2b = (const float*)d_in[4];
    const float* f1w = (const float*)d_in[5];
    const float* f1b = (const float*)d_in[6];
    const float* f2w = (const float*)d_in[7];
    const float* f2b = (const float*)d_in[8];
    float* out = (float*)d_out;

    char* ws = (char*)d_ws;
    // workspace layout (bytes); peak ~44 MB
    float* mm    = (float*)(ws + 0);            // 12 f32
    float* Weff1 = (float*)(ws + 256);          // 500 f32
    float* Beff1 = (float*)(ws + 2304);         // 20 f32
    float* Beff2 = (float*)(ws + 2432);         // 64 f32
    float* Beff3 = (float*)(ws + 2688);         // 512 f32
    h16*   Bt2   = (h16*)  (ws + 8192);         // 64*512 f16 = 64KB
    h16*   Bt3   = (h16*)  (ws + 131072);       // 512*800 f16 = 800KB
    h16*   Xq    = (h16*)  (ws + (1u << 20));   // 4096*784 f16 = 6.1MB
    h16*   P1    = (h16*)  (ws + (8u << 20));   // 4096*20*144 f16 = 22.5MB
    h16*   P2    = (h16*)  (ws + (32u << 20));  // 4096*800 f16 = 6.25MB
    h16*   Q3    = (h16*)  (ws + (40u << 20));  // 4096*512 f16 = 4MB

    // 1) param min/max + effective weight/bias prep
    k_minmax<<<6, 256, 0, stream>>>(c1w, 500, c1b, 20, c2w, 25000,
                                    c2b, 50, f1w, 400000, f1b, 500, mm);
    k_prep_small<<<4, 256, 0, stream>>>(c1w, c1b, c2b, f1b, mm,
                                        Weff1, Beff1, Beff2, Beff3);
    k_prep_bt2<<<(64 * 512) / 256, 256, 0, stream>>>(c2w, mm, Bt2);
    k_prep_bt3<<<(512 * 800) / 256, 256, 0, stream>>>(f1w, mm, Bt3);

    // 2) quantize input (X = q - zp, exact integers in fp16)
    int nin = BATCH * 784;
    k_quant_input<<<(nin + 255) / 256, 256, 0, stream>>>(x, Xq, nin);

    // 3) conv1 + requant + pool (fused)
    k_conv1_pool<<<(BATCH * 144) / 256, 256, 0, stream>>>(Xq, Weff1, Beff1, P1);

    // 4) conv2 WMMA + requant + pool + flatten (fused, one block per image)
    k_conv2_wmma<<<BATCH, 512, 0, stream>>>(P1, Bt2, Beff2, P2);

    // 5) fc1 WMMA GEMM + requant
    k_fc1_wmma<<<dim3(BATCH / 64, 16), 256, 0, stream>>>(P2, Bt3, Beff3, Q3);

    // 6) dequant + fc2 + log_softmax
    k_fc2_lsm<<<BATCH / 8, 256, 0, stream>>>(Q3, f2w, f2b, out);
}
